// GTNet_Transformer_61452392071961
// MI455X (gfx1250) — matrix operand order repaired and checked by reading, not compiled, not run on hardware
//
#include <hip/hip_runtime.h>
#include <hip/hip_fp16.h>

typedef __attribute__((ext_vector_type(16))) _Float16 v16h;
typedef __attribute__((ext_vector_type(8)))  float    v8f;
typedef int v4i __attribute__((vector_size(16)));   // matches async-LDS builtin param

union H16u { v16h v; _Float16 h[16]; uint4 u[2]; };

constexpr int kHW   = 16384;   // 128*128 pixels
constexpr int kCIN  = 1024;    // input channels
constexpr int kDTOT = 512;     // total model dim
constexpr int kDM   = 256;     // per-head dim
constexpr int kPA   = 2048;    // query rows

#define GLOBAL_AS __attribute__((address_space(1)))
#define LDS_AS    __attribute__((address_space(3)))

// 16-byte global -> LDS copy: async DMA (GLOBAL_LOAD_ASYNC_TO_LDS_B128) on gfx1250.
__device__ __forceinline__ void cp_async16(void* lds, const void* g) {
#if __has_builtin(__builtin_amdgcn_global_load_async_to_lds_b128)
  __builtin_amdgcn_global_load_async_to_lds_b128(
      (GLOBAL_AS v4i*)g, (LDS_AS v4i*)lds, 0, 0);
#else
  *(uint4*)lds = *(const uint4*)g;
#endif
}

__device__ __forceinline__ void cp_async_wait() {
#if __has_builtin(__builtin_amdgcn_global_load_async_to_lds_b128)
#if __has_builtin(__builtin_amdgcn_s_wait_asynccnt)
  __builtin_amdgcn_s_wait_asynccnt(0);
#else
  asm volatile("s_wait_asynccnt 0x0" ::: "memory");
#endif
#endif
}

__device__ __forceinline__ v16h load_frag16(const _Float16* lo, const _Float16* hi) {
  H16u f;
  f.u[0] = *(const uint4*)lo;
  f.u[1] = *(const uint4*)hi;
  return f.v;
}

__device__ __forceinline__ v8f wmma_f16(v16h a, v16h b, v8f c) {
  return __builtin_amdgcn_wmma_f32_16x16x32_f16(false, a, false, b, (short)0, c, false, false);
}

// ---------------------------------------------------------------- converts
__global__ void cvt_f32_to_f16(const float* __restrict__ in,
                               _Float16* __restrict__ out, int n) {
  int i = blockIdx.x * blockDim.x + threadIdx.x;
  int stride = gridDim.x * blockDim.x;
  for (; i < n; i += stride) out[i] = (_Float16)in[i];
}

// f_map (C x P fp32) -> fmap_t (P x C f16); 32x32 LDS tiles, coalesced both sides
__global__ void __launch_bounds__(256) transpose_cvt(
    const float* __restrict__ in, _Float16* __restrict__ out) {
  __shared__ _Float16 tile[32][33];
  const int c0 = blockIdx.y * 32;
  const int p0 = blockIdx.x * 32;
  const int lx = threadIdx.x & 31, ly = threadIdx.x >> 5;  // 32 x 8
#pragma unroll
  for (int i = 0; i < 32; i += 8)
    tile[ly + i][lx] = (_Float16)in[(size_t)(c0 + ly + i) * kHW + p0 + lx];
  __syncthreads();
#pragma unroll
  for (int i = 0; i < 32; i += 8)
    out[(size_t)(p0 + ly + i) * kCIN + c0 + lx] = tile[lx][ly + i];
}

// ---------------------------------------------------------------- projection
// O[o,p] = relu(BN(sum_k W[o,k] * Xt[p,k]))  ; W:512x1024 f16, Xt:16384x1024 f16
// transposed==1 -> out[p*512+o] (K transposed, pixel-major: feeds S-GEMM B tiles)
// transposed==0 -> out[o*HW+p]  (V plain, channel-major: feeds AV-GEMM B tiles)
__global__ void __launch_bounds__(256) proj_bn_relu(
    const _Float16* __restrict__ Wh, const _Float16* __restrict__ Xt,
    const float* __restrict__ bg, const float* __restrict__ bb,
    const float* __restrict__ bm, const float* __restrict__ bv,
    _Float16* __restrict__ out, int transposed)
{
  __shared__ _Float16 sW[2][128][40];  // [m][k], 16B-aligned padded rows
  __shared__ _Float16 sX[2][128][40];  // [n][k]
  const int tid  = threadIdx.x;
  const int w    = tid >> 5, l = tid & 31;
  const int ml   = l & 15,  half = l >> 4;
  const int m0   = blockIdx.y * 128;
  const int p0   = blockIdx.x * 128;
  const int m_off = (w & 1) * 64;
  const int n_off = (w >> 1) * 32;

  auto stage = [&](int kk, int buf) {
    for (int c2 = tid; c2 < 512; c2 += 256) {
      int m = c2 >> 2, k8 = (c2 & 3) * 8;
      cp_async16(&sW[buf][m][k8], Wh + (size_t)(m0 + m) * kCIN + kk + k8);
    }
    for (int c2 = tid; c2 < 512; c2 += 256) {
      int n = c2 >> 2, k8 = (c2 & 3) * 8;
      cp_async16(&sX[buf][n][k8], Xt + (size_t)(p0 + n) * kCIN + kk + k8);
    }
  };

  const v8f z8 = {0.f,0.f,0.f,0.f,0.f,0.f,0.f,0.f};
  v8f acc[4][2];
#pragma unroll
  for (int mi = 0; mi < 4; ++mi)
#pragma unroll
    for (int ni = 0; ni < 2; ++ni) acc[mi][ni] = z8;

  stage(0, 0);
  cp_async_wait();
  __syncthreads();

  for (int kk = 0; kk < kCIN; kk += 32) {
    const int buf = (kk >> 5) & 1;
    if (kk + 32 < kCIN) stage(kk + 32, buf ^ 1);

    v16h afr[4], bfr[2];
#pragma unroll
    for (int mi = 0; mi < 4; ++mi) {
      int row = m_off + mi * 16 + ml;
      afr[mi] = load_frag16(&sW[buf][row][half * 8], &sW[buf][row][16 + half * 8]);
    }
#pragma unroll
    for (int ni = 0; ni < 2; ++ni) {
      int col = n_off + ni * 16 + ml;
      bfr[ni] = load_frag16(&sX[buf][col][half * 8], &sX[buf][col][16 + half * 8]);
    }
#pragma unroll
    for (int mi = 0; mi < 4; ++mi)
#pragma unroll
      for (int ni = 0; ni < 2; ++ni)
        acc[mi][ni] = wmma_f16(afr[mi], bfr[ni], acc[mi][ni]);

    if (kk + 32 < kCIN) cp_async_wait();
    __syncthreads();
  }

  // epilogue: batch-norm (inference form) + relu, store f16
#pragma unroll
  for (int mi = 0; mi < 4; ++mi) {
#pragma unroll
    for (int r = 0; r < 8; ++r) {
      int o = m0 + m_off + mi * 16 + half * 8 + r;
      float g = bg[o], b = bb[o], mu = bm[o];
      float rs = rsqrtf(bv[o] + 1e-5f);
#pragma unroll
      for (int ni = 0; ni < 2; ++ni) {
        int p = p0 + n_off + ni * 16 + ml;
        float x = acc[mi][ni][r];
        float y = fmaxf((x - mu) * rs * g + b, 0.f);
        if (transposed) out[(size_t)p * kDTOT + o] = (_Float16)y;
        else            out[(size_t)o * kHW  + p] = (_Float16)y;
      }
    }
  }
}

// ---------------------------------------------------------------- flash attention
// A[row, s+chan] = softmax(qh @ k0 / 16) @ v0^T
// kt: pixel-major K (HW x 512), vf: channel-major V (512 x HW); both L2-resident.
// Double-buffered async LDS tiles, online softmax, all GEMMs on v_wmma.
__global__ void __launch_bounds__(256) attn_flash(
    const _Float16* __restrict__ qh, const _Float16* __restrict__ kt,
    const _Float16* __restrict__ vf, float* __restrict__ Abuf)
{
  __shared__ _Float16 sK[2][32][264];   // [key][chan], chan contiguous
  __shared__ _Float16 sV[2][256][40];   // [chan][key], key contiguous
  __shared__ _Float16 sP[8][16][32];    // per-wave P bounce buffer
  const int tid  = threadIdx.x;
  const int w    = tid >> 5, l = tid & 31;
  const int ml   = l & 15,  half = l >> 4;
  const int head = blockIdx.y;
  const int s    = head * kDM;
  const int qrow0 = blockIdx.x * 128 + w * 16;

  auto stage = [&](int t, int buf) {
    const int p0 = t * 32;
    for (int c2 = tid; c2 < 1024; c2 += 256) {
      int j = c2 >> 5, c8 = (c2 & 31) * 8;
      cp_async16(&sK[buf][j][c8], kt + (size_t)(p0 + j) * kDTOT + s + c8);
    }
    for (int c2 = tid; c2 < 1024; c2 += 256) {
      int chan = c2 >> 2, j8 = (c2 & 3) * 8;
      cp_async16(&sV[buf][chan][j8], vf + (size_t)(s + chan) * kHW + p0 + j8);
    }
  };

  // q fragments for 16 rows x 256 channels (8 K-chunks of 32)
  v16h aq[8];
#pragma unroll
  for (int k8 = 0; k8 < 8; ++k8) {
    const _Float16* base = qh + (size_t)(qrow0 + ml) * kDTOT + s + k8 * 32 + half * 8;
    aq[k8] = load_frag16(base, base + 16);
  }

  const v8f z8 = {0.f,0.f,0.f,0.f,0.f,0.f,0.f,0.f};
  v8f oacc[16];
#pragma unroll
  for (int ni = 0; ni < 16; ++ni) oacc[ni] = z8;
  float mrow[8], lrow[8];
#pragma unroll
  for (int r = 0; r < 8; ++r) { mrow[r] = -1e30f; lrow[r] = 0.f; }

  stage(0, 0);
  cp_async_wait();
  __syncthreads();

  constexpr int NT = kHW / 32;
  for (int t = 0; t < NT; ++t) {
    const int buf = t & 1;
    if (t + 1 < NT) stage(t + 1, buf ^ 1);

    // S tile = qh_tile @ K_tile (16x32), 8 K-steps
    v8f sa0 = z8, sa1 = z8;
#pragma unroll
    for (int k8 = 0; k8 < 8; ++k8) {
      v16h b0 = load_frag16(&sK[buf][ml][k8 * 32 + half * 8],
                            &sK[buf][ml][k8 * 32 + 16 + half * 8]);
      v16h b1 = load_frag16(&sK[buf][16 + ml][k8 * 32 + half * 8],
                            &sK[buf][16 + ml][k8 * 32 + 16 + half * 8]);
      sa0 = wmma_f16(aq[k8], b0, sa0);
      sa1 = wmma_f16(aq[k8], b1, sa1);
    }
    sa0 *= 0.0625f;  // 1/sqrt(256)
    sa1 *= 0.0625f;

    // online softmax per row r (row = half*8 + r, spread over 16 lanes)
#pragma unroll
    for (int r = 0; r < 8; ++r) {
      float mloc = fmaxf(sa0[r], sa1[r]);
#pragma unroll
      for (int off = 8; off >= 1; off >>= 1)
        mloc = fmaxf(mloc, __shfl_xor(mloc, off, 32));
      float mnew = fmaxf(mrow[r], mloc);
      float sc   = __expf(mrow[r] - mnew);
      float p0v  = __expf(sa0[r] - mnew);
      float p1v  = __expf(sa1[r] - mnew);
      float sl   = p0v + p1v;
#pragma unroll
      for (int off = 8; off >= 1; off >>= 1) sl += __shfl_xor(sl, off, 32);
      lrow[r] = lrow[r] * sc + sl;
      mrow[r] = mnew;
      int row = half * 8 + r;
      sP[w][row][ml]      = (_Float16)p0v;
      sP[w][row][16 + ml] = (_Float16)p1v;
#pragma unroll
      for (int ni = 0; ni < 16; ++ni) oacc[ni][r] *= sc;
    }

    // P (C-layout) -> A-fragment via per-wave LDS bounce (wave LDS ops in-order)
    v16h ap = load_frag16(&sP[w][ml][half * 8], &sP[w][ml][16 + half * 8]);
#pragma unroll
    for (int ni = 0; ni < 16; ++ni) {
      v16h bv = load_frag16(&sV[buf][ni * 16 + ml][half * 8],
                            &sV[buf][ni * 16 + ml][16 + half * 8]);
      oacc[ni] = wmma_f16(ap, bv, oacc[ni]);
    }

    if (t + 1 < NT) cp_async_wait();
    __syncthreads();
  }

  // finalize: divide by row sum, write fp32
#pragma unroll
  for (int ni = 0; ni < 16; ++ni)
#pragma unroll
    for (int r = 0; r < 8; ++r) {
      int row  = qrow0 + half * 8 + r;
      int chan = ni * 16 + ml;
      Abuf[(size_t)row * kDTOT + s + chan] = oacc[ni][r] / lrow[r];
    }
}

// ---------------------------------------------------------------- FF + LayerNorms
__global__ void __launch_bounds__(256) ffn_norm(
    const float* __restrict__ Abuf, const float* __restrict__ qf,
    const float* __restrict__ n1a, const float* __restrict__ n1b,
    const float* __restrict__ n2a, const float* __restrict__ n2b,
    const float* __restrict__ w1, const float* __restrict__ b1,
    const float* __restrict__ w2, const float* __restrict__ b2,
    _Float16* __restrict__ fh, int jbase)
{
  __shared__ float sQ[8][256];
  __shared__ float sH[8][128];
  const int tid = threadIdx.x, w = tid >> 5, l = tid & 31;
  const int R = blockIdx.x * 8 + w;
  const int head = blockIdx.y, s = head * kDM;
  const int flag = jbase + head;

  float x[8];
  float sum = 0.f;
#pragma unroll
  for (int i = 0; i < 8; ++i) {
    int c = l + i * 32;
    x[i] = Abuf[(size_t)R * kDTOT + s + c] + qf[(size_t)R * kDTOT + s + c];
    sum += x[i];
  }
#pragma unroll
  for (int off = 16; off >= 1; off >>= 1) sum += __shfl_xor(sum, off, 32);
  float mu = sum * (1.f / 256.f);
  float vs = 0.f;
#pragma unroll
  for (int i = 0; i < 8; ++i) { float d = x[i] - mu; vs += d * d; }
#pragma unroll
  for (int off = 16; off >= 1; off >>= 1) vs += __shfl_xor(vs, off, 32);
  float inv1 = 1.f / (sqrtf(vs * (1.f / 255.f)) + 1e-6f);  // ddof=1

  float qn[8];
#pragma unroll
  for (int i = 0; i < 8; ++i) {
    int c = l + i * 32;
    qn[i] = n1a[flag * 256 + c] * (x[i] - mu) * inv1 + n1b[flag * 256 + c];
    sQ[w][c] = qn[i];
  }
  __syncthreads();

#pragma unroll
  for (int jj = 0; jj < 4; ++jj) {
    int j = l + jj * 32;
    float a = b1[flag * 128 + j];
    const float* wr = w1 + ((size_t)flag * 128 + j) * 256;
    for (int k = 0; k < 256; ++k) a += sQ[w][k] * wr[k];
    sH[w][j] = fmaxf(a, 0.f);
  }
  __syncthreads();

  float y[8]; float sum2 = 0.f;
#pragma unroll
  for (int i = 0; i < 8; ++i) {
    int c = l + i * 32;
    float a = b2[flag * 256 + c];
    const float* wr = w2 + ((size_t)flag * 256 + c) * 128;
    for (int k = 0; k < 128; ++k) a += sH[w][k] * wr[k];
    y[i] = qn[i] + fmaxf(a, 0.f);
    sum2 += y[i];
  }
#pragma unroll
  for (int off = 16; off >= 1; off >>= 1) sum2 += __shfl_xor(sum2, off, 32);
  float mu2 = sum2 * (1.f / 256.f);
  float vs2 = 0.f;
#pragma unroll
  for (int i = 0; i < 8; ++i) { float d = y[i] - mu2; vs2 += d * d; }
#pragma unroll
  for (int off = 16; off >= 1; off >>= 1) vs2 += __shfl_xor(vs2, off, 32);
  float inv2 = 1.f / (sqrtf(vs2 * (1.f / 255.f)) + 1e-6f);
#pragma unroll
  for (int i = 0; i < 8; ++i) {
    int c = l + i * 32;
    float f = n2a[flag * 256 + c] * (y[i] - mu2) * inv2 + n2b[flag * 256 + c];
    fh[(size_t)R * kDTOT + s + c] = (_Float16)f;
  }
}

// ---------------------------------------------------------------- combine GEMM
__global__ void __launch_bounds__(256) comb_wmma(
    const _Float16* __restrict__ fh, const _Float16* __restrict__ wc,
    const float* __restrict__ cb, const float* __restrict__ ag,
    float* __restrict__ qout, _Float16* __restrict__ qhout)
{
  const int tid = threadIdx.x, w = tid >> 5, l = tid & 31;
  const int ml = l & 15, half = l >> 4;
  const int M0 = blockIdx.x * 256, N0 = blockIdx.y * 128;
  const int m_off = (w & 3) * 64, n_off = (w >> 2) * 64;

  const v8f z8 = {0.f,0.f,0.f,0.f,0.f,0.f,0.f,0.f};
  v8f acc[4][4];
#pragma unroll
  for (int mi = 0; mi < 4; ++mi)
#pragma unroll
    for (int ni = 0; ni < 4; ++ni) acc[mi][ni] = z8;

  for (int kk = 0; kk < kDTOT; kk += 32) {
    if (kk + 32 < kDTOT)
      __builtin_prefetch(fh + (size_t)(M0 + m_off + ml) * kDTOT + kk + 32, 0, 0);
    v16h afr[4], bfr[4];
#pragma unroll
    for (int mi = 0; mi < 4; ++mi) {
      const _Float16* base = fh + (size_t)(M0 + m_off + mi * 16 + ml) * kDTOT + kk + half * 8;
      afr[mi] = load_frag16(base, base + 16);
    }
#pragma unroll
    for (int ni = 0; ni < 4; ++ni) {
      const _Float16* base = wc + (size_t)(N0 + n_off + ni * 16 + ml) * kDTOT + kk + half * 8;
      bfr[ni] = load_frag16(base, base + 16);
    }
#pragma unroll
    for (int mi = 0; mi < 4; ++mi)
#pragma unroll
      for (int ni = 0; ni < 4; ++ni)
        acc[mi][ni] = wmma_f16(afr[mi], bfr[ni], acc[mi][ni]);
  }

#pragma unroll
  for (int mi = 0; mi < 4; ++mi)
#pragma unroll
    for (int r = 0; r < 8; ++r) {
      int row = M0 + m_off + mi * 16 + half * 8 + r;
#pragma unroll
      for (int ni = 0; ni < 4; ++ni) {
        int n = N0 + n_off + ni * 16 + ml;
        float y = fmaxf(acc[mi][ni][r] + cb[n], 0.f);
        y *= ag[(size_t)row * kDTOT + n];
        qout[(size_t)row * kDTOT + n]  = y;
        qhout[(size_t)row * kDTOT + n] = (_Float16)y;
      }
    }
}

// ---------------------------------------------------------------- host launcher
extern "C" void kernel_launch(void* const* d_in, const int* in_sizes, int n_in,
                              void* d_out, int out_size, void* d_ws, size_t ws_size,
                              hipStream_t stream)
{
  const float* f_map  = (const float*)d_in[0];
  const float* q      = (const float*)d_in[1];
  const float* att_g  = (const float*)d_in[2];
  const float* wk     = (const float*)d_in[3];
  const float* bn_k_g = (const float*)d_in[4];
  const float* bn_k_b = (const float*)d_in[5];
  const float* bn_k_m = (const float*)d_in[6];
  const float* bn_k_v = (const float*)d_in[7];
  const float* wv     = (const float*)d_in[8];
  const float* bn_v_g = (const float*)d_in[9];
  const float* bn_v_b = (const float*)d_in[10];
  const float* bn_v_m = (const float*)d_in[11];
  const float* bn_v_v = (const float*)d_in[12];
  const float* tx_n1a = (const float*)d_in[13];
  const float* tx_n1b = (const float*)d_in[14];
  const float* tx_n2a = (const float*)d_in[15];
  const float* tx_n2b = (const float*)d_in[16];
  const float* tx_w1  = (const float*)d_in[17];
  const float* tx_b1  = (const float*)d_in[18];
  const float* tx_w2  = (const float*)d_in[19];
  const float* tx_b2  = (const float*)d_in[20];
  const float* comb_w = (const float*)d_in[21];
  const float* comb_b = (const float*)d_in[22];

  char* ws = (char*)d_ws;
  size_t off = 0;
  auto carve = [&](size_t bytes) {
    void* p = ws + off;
    off += (bytes + 255) & ~(size_t)255;
    return p;
  };
  _Float16* fmap_t  = (_Float16*)carve((size_t)kHW * kCIN * 2);    // pixel-major f16
  _Float16* wk_h    = (_Float16*)carve((size_t)kDTOT * kCIN * 2);
  _Float16* wv_h    = (_Float16*)carve((size_t)kDTOT * kCIN * 2);
  _Float16* kt_h    = (_Float16*)carve((size_t)kHW * kDTOT * 2);   // K transposed (L2-resident)
  _Float16* vf_h    = (_Float16*)carve((size_t)kDTOT * kHW * 2);   // V plain     (L2-resident)
  _Float16* q_h     = (_Float16*)carve((size_t)kPA * kDTOT * 2);
  _Float16* combw_h = (_Float16*)carve((size_t)2 * kDTOT * kDTOT * 2);
  _Float16* f_h     = (_Float16*)carve((size_t)kPA * kDTOT * 2);
  float*    A_buf   = (float*)carve((size_t)kPA * kDTOT * 4);
  float*    q_cur   = (float*)carve((size_t)kPA * kDTOT * 4);
  (void)ws_size; (void)in_sizes; (void)n_in; (void)out_size;

  // operand conversion (f_map transposed to pixel-major so all later LDS
  // staging is straight 16B async copies)
  transpose_cvt<<<dim3(kHW / 32, kCIN / 32), 256, 0, stream>>>(f_map, fmap_t);
  cvt_f32_to_f16<<<512, 256, 0, stream>>>(wk, wk_h, kDTOT * kCIN);
  cvt_f32_to_f16<<<512, 256, 0, stream>>>(wv, wv_h, kDTOT * kCIN);
  cvt_f32_to_f16<<<512, 256, 0, stream>>>(q, q_h, kPA * kDTOT);
  cvt_f32_to_f16<<<512, 256, 0, stream>>>(comb_w, combw_h, 2 * kDTOT * kDTOT);

  // K (transposed out) / V (plain out) projections: WMMA GEMM + fused BN + ReLU
  proj_bn_relu<<<dim3(kHW / 128, kDTOT / 128), 256, 0, stream>>>(
      wk_h, fmap_t, bn_k_g, bn_k_b, bn_k_m, bn_k_v, kt_h, 1);
  proj_bn_relu<<<dim3(kHW / 128, kDTOT / 128), 256, 0, stream>>>(
      wv_h, fmap_t, bn_v_g, bn_v_b, bn_v_m, bn_v_v, vf_h, 0);

  const float* qsrc = q;
  for (int j = 0; j < 2; ++j) {
    attn_flash<<<dim3(kPA / 128, 2), 256, 0, stream>>>(q_h, kt_h, vf_h, A_buf);
    ffn_norm<<<dim3(kPA / 8, 2), 256, 0, stream>>>(
        A_buf, qsrc, tx_n1a, tx_n1b, tx_n2a, tx_n2b,
        tx_w1, tx_b1, tx_w2, tx_b2, f_h, j * 2);
    float* dst = (j == 1) ? (float*)d_out : q_cur;
    comb_wmma<<<dim3(kPA / 256, kDTOT / 128), 256, 0, stream>>>(
        f_h, combw_h + (size_t)j * kDTOT * kDTOT, comb_b + j * kDTOT,
        att_g, dst, q_h);
    qsrc = dst;
  }
}
